// SparseMoE_62886911148312
// MI455X (gfx1250) — matrix operand order repaired and checked by reading, not compiled, not run on hardware
//
#include <hip/hip_runtime.h>
#include <hip/hip_bf16.h>
#include <math.h>

// ---------------- problem constants (from reference) ----------------
#define E_    8
#define DIM_  256
#define TDIM_ 768
#define HID_  1024
#define NTOK_ 16384         // B*H*W = 16*32*32
#define PIX_  1024          // H*W

typedef __attribute__((ext_vector_type(16))) _Float16 v16h;
typedef __attribute__((ext_vector_type(8)))  float    v8f;

union Frag16 { v16h h; uint4 u[2]; };

__device__ __forceinline__ float softplus_f(float x) {
  return (x > 20.f) ? x : log1pf(__expf(x));
}

// ---------------- 1) sample Bayesian router weights (elementwise) ----------------
__global__ void sample_weights(const float* __restrict__ fmu, const float* __restrict__ frho, const float* __restrict__ ef,
                               const float* __restrict__ tmu, const float* __restrict__ trho, const float* __restrict__ et,
                               const float* __restrict__ cmu, const float* __restrict__ crho, const float* __restrict__ ec,
                               float* __restrict__ fw, float* __restrict__ tw, float* __restrict__ cw) {
  const int NF = DIM_ * 128, NT = TDIM_ * 128, NC = 256 * E_;
  int i = blockIdx.x * 256 + threadIdx.x;
  if (i < NF) { fw[i] = fmu[i] + softplus_f(frho[i]) * ef[i]; return; }
  int j = i - NF;
  if (j < NT) { tw[j] = tmu[j] + softplus_f(trho[j]) * et[j]; return; }
  int k = j - NT;
  if (k < NC) { cw[k] = cmu[k] + softplus_f(crho[k]) * ec[k]; }
}

// ---------------- 2) tiled transpose + f32->f16 convert ----------------
// in:  [batch][R][C] f32   ->   out: [batch][C][R] f16
__global__ void transpose_f32_f16(const float* __restrict__ in, _Float16* __restrict__ out,
                                  int R, int C) {
  __shared__ float tile[32][33];
  long base = (long)blockIdx.z * R * C;
  const float* ip = in + base;
  _Float16* op = out + base;
  int c0 = blockIdx.x * 32, r0 = blockIdx.y * 32;
  int tx = threadIdx.x, ty = threadIdx.y;
  for (int i = ty; i < 32; i += 8)
    tile[i][tx] = ip[(long)(r0 + i) * C + c0 + tx];
  __syncthreads();
  for (int i = ty; i < 32; i += 8)
    op[(long)(c0 + i) * R + r0 + tx] = (_Float16)tile[tx][i];
}

// ---------------- 3) per-image text projection: tcomb[16][128] = w @ tw ----------------
__global__ void tcomb_kernel(const float* __restrict__ w, const float* __restrict__ tw,
                             float* __restrict__ tcomb) {
  __shared__ float wr[TDIM_];
  int img = blockIdx.x, j = threadIdx.x;  // 128 threads
  for (int k = j; k < TDIM_; k += 128) wr[k] = w[img * TDIM_ + k];
  __syncthreads();
  float acc = 0.f;
  for (int k = 0; k < TDIM_; ++k) acc += wr[k] * tw[k * 128 + j];
  tcomb[img * 128 + j] = acc;
}

// ---------------- 4) router: x@fw | concat | @cw | /temp | softmax -> probs ----------------
// 16 tokens per 128-thread block (all same image since 1024 tokens/image)
__global__ void router_kernel(const _Float16* __restrict__ xh, const float* __restrict__ fw,
                              const float* __restrict__ cw, const float* __restrict__ tcomb,
                              const float* __restrict__ temp, float* __restrict__ probs) {
  __shared__ float xs[16][DIM_];
  __shared__ float c1[16][128];
  __shared__ float tc[128];
  __shared__ float lg[16][E_];
  int n0 = blockIdx.x * 16;
  int img = n0 >> 10;
  int tid = threadIdx.x;
  for (int idx = tid; idx < 16 * DIM_; idx += 128) {
    int t = idx >> 8, k = idx & 255;
    xs[t][k] = (float)xh[(long)(n0 + t) * DIM_ + k];
  }
  tc[tid] = tcomb[img * 128 + tid];
  __syncthreads();
  float acc[16];
#pragma unroll
  for (int t = 0; t < 16; ++t) acc[t] = 0.f;
  for (int k = 0; k < DIM_; ++k) {
    float fv = fw[k * 128 + tid];        // coalesced across the 128 threads
#pragma unroll
    for (int t = 0; t < 16; ++t) acc[t] += xs[t][k] * fv;
  }
  for (int t = 0; t < 16; ++t) c1[t][tid] = acc[t];
  __syncthreads();
  int t = tid >> 3, e = tid & 7;         // 16 tokens x 8 experts
  float lacc = 0.f;
  for (int j = 0; j < 128; ++j) lacc += c1[t][j] * cw[j * E_ + e];
  for (int j = 0; j < 128; ++j) lacc += tc[j] * cw[(128 + j) * E_ + e];
  float tclip = fmaxf(temp[0], 0.1f);
  lg[t][e] = lacc / tclip;
  __syncthreads();
  float m = lg[t][0];
#pragma unroll
  for (int q = 1; q < 8; ++q) m = fmaxf(m, lg[t][q]);
  float s = 0.f;
#pragma unroll
  for (int q = 0; q < 8; ++q) s += __expf(lg[t][q] - m);
  probs[(long)(n0 + t) * E_ + e] = __expf(lg[t][e] - m) / s;
}

// ---------------- 5) KL divergence (single block, deterministic reduction) ----------------
__global__ void kl_kernel(const float* __restrict__ fmu, const float* __restrict__ frho,
                          const float* __restrict__ tmu, const float* __restrict__ trho,
                          const float* __restrict__ cmu, const float* __restrict__ crho,
                          float* __restrict__ out_kl) {
  __shared__ float red[256];
  int tid = threadIdx.x;
  float s = 0.f;
  const int NF = DIM_ * 128, NT = TDIM_ * 128, NC = 256 * E_;
  for (int i = tid; i < NF; i += 256) { float sp = softplus_f(frho[i]); float mu = fmu[i];
    s += 0.5f * (-2.f * __logf(sp) + sp * sp + mu * mu - 1.f); }
  for (int i = tid; i < NT; i += 256) { float sp = softplus_f(trho[i]); float mu = tmu[i];
    s += 0.5f * (-2.f * __logf(sp) + sp * sp + mu * mu - 1.f); }
  for (int i = tid; i < NC; i += 256) { float sp = softplus_f(crho[i]); float mu = cmu[i];
    s += 0.5f * (-2.f * __logf(sp) + sp * sp + mu * mu - 1.f); }
  red[tid] = s; __syncthreads();
  for (int st = 128; st > 0; st >>= 1) { if (tid < st) red[tid] += red[tid + st]; __syncthreads(); }
  if (tid == 0) out_kl[0] = 1e-4f * red[0];
}

// ---------------- 6) fused dense MoE expert pass (WMMA core) ----------------
// 64 tokens per 256-thread (8-wave) workgroup; HID processed in chunks of 128.
// out[n,:] = sum_e p[n,e]*(gelu(x W1e + b1e) W2e + b2e); probs folded into h before GEMM2.
#define MT 64
#define HC 128

__global__ void __launch_bounds__(256)
moe_expert_kernel(const _Float16* __restrict__ xh,   // [N][DIM]
                  const _Float16* __restrict__ W1t,  // [E][HID][DIM] (transposed: B cols contiguous)
                  const _Float16* __restrict__ W2t,  // [E][DIM][HID]
                  const float* __restrict__ b1,      // [E][HID]
                  const float* __restrict__ b2,      // [E][DIM]
                  const float* __restrict__ probs,   // [N][E]
                  float* __restrict__ out)           // [B][DIM][PIX]
{
  __shared__ _Float16 xs[MT][DIM_ + 8];  // +8 halves pad vs bank conflicts
  __shared__ _Float16 hs[MT][HC + 8];
  __shared__ float    ps[MT][E_];
  int tid = threadIdx.x;
  int wave = tid >> 5, lane = tid & 31;
  int lr = lane & 15, lh = lane >> 4;
  long n0 = (long)blockIdx.x * MT;

  // stage x tile (f16, row-major) and this block's probs
  for (int idx = tid; idx < MT * (DIM_ / 2); idx += 256) {
    int row = idx >> 7, c2 = idx & 127;  // 128 uints per row
    const unsigned int* src = (const unsigned int*)(xh + (n0 + row) * DIM_);
    *(unsigned int*)&xs[row][c2 * 2] = src[c2];
  }
  for (int idx = tid; idx < MT * E_; idx += 256)
    ps[idx >> 3][idx & 7] = probs[(n0 + (idx >> 3)) * E_ + (idx & 7)];
  __syncthreads();

  v8f acc[4][2] = {};   // GEMM2 accumulators: 4 row-tiles x 2 col-tiles per wave

  for (int e = 0; e < E_; ++e) {
    for (int ch = 0; ch < HID_ / HC; ++ch) {
      // ---- GEMM1: wave owns h columns [ch*HC + wave*16 .. +15], all 64 rows ----
      int hcol = ch * HC + wave * 16 + lr;
      const _Float16* wbase = W1t + ((long)e * HID_ + hcol) * DIM_;
      Frag16 bf[8];
#pragma unroll
      for (int ks = 0; ks < 8; ++ks) {   // K = 256 = 8 x 32
        int off = ks * 32 + lh * 8;      // B lane layout: col=lane&15, k in [8*lh,+7] and [8*lh+16,+23]
        bf[ks].u[0] = *(const uint4*)(wbase + off);
        bf[ks].u[1] = *(const uint4*)(wbase + off + 16);
      }
      float b1v = b1[e * HID_ + hcol];
#pragma unroll
      for (int rt = 0; rt < 4; ++rt) {
        v8f hacc = {};
#pragma unroll
        for (int ks = 0; ks < 8; ++ks) {
          Frag16 af;                      // A lane layout: row=lane&15, k in [8*lh,+7] and [8*lh+16,+23]
          int row = rt * 16 + lr;
          int k0 = ks * 32 + lh * 8;
          af.u[0] = *(const uint4*)&xs[row][k0];
          af.u[1] = *(const uint4*)&xs[row][k0 + 16];
          hacc = __builtin_amdgcn_wmma_f32_16x16x32_f16(false, af.h, false, bf[ks].h,
                                                        (short)0, hacc, false, false);
        }
        // fused bias + exact GELU + prob scale, store f16 h-tile to LDS
#pragma unroll
        for (int i = 0; i < 8; ++i) {
          int row = rt * 16 + lh * 8 + i;  // C/D layout: vgpr i -> M = i + 8*lh
          float v = hacc[i] + b1v;
          v = 0.5f * v * (1.f + erff(v * 0.70710678118f));
          v *= ps[row][e];
          hs[row][wave * 16 + lr] = (_Float16)v;
        }
      }
      __syncthreads();
      // ---- GEMM2: acc += hs @ W2t[e][:, ch*HC..] ; wave owns d col-tiles {2w, 2w+1} ----
      Frag16 bg[2][4];
#pragma unroll
      for (int ctl = 0; ctl < 2; ++ctl) {
        int d = (wave * 2 + ctl) * 16 + lr;
        const _Float16* w2base = W2t + ((long)e * DIM_ + d) * HID_ + ch * HC;
#pragma unroll
        for (int ks = 0; ks < 4; ++ks) {  // K = HC = 4 x 32
          int off = ks * 32 + lh * 8;
          bg[ctl][ks].u[0] = *(const uint4*)(w2base + off);
          bg[ctl][ks].u[1] = *(const uint4*)(w2base + off + 16);
        }
      }
#pragma unroll
      for (int rt = 0; rt < 4; ++rt) {
#pragma unroll
        for (int ks = 0; ks < 4; ++ks) {
          Frag16 af;
          int row = rt * 16 + lr;
          int k0 = ks * 32 + lh * 8;
          af.u[0] = *(const uint4*)&hs[row][k0];
          af.u[1] = *(const uint4*)&hs[row][k0 + 16];
#pragma unroll
          for (int ctl = 0; ctl < 2; ++ctl)
            acc[rt][ctl] = __builtin_amdgcn_wmma_f32_16x16x32_f16(false, af.h, false, bg[ctl][ks].h,
                                                                  (short)0, acc[rt][ctl], false, false);
        }
      }
      __syncthreads();  // before next chunk overwrites hs
    }
  }

  // epilogue: += sum_e p[n,e]*b2[e,d]; scatter to NCHW output
#pragma unroll
  for (int ctl = 0; ctl < 2; ++ctl) {
    int d = (wave * 2 + ctl) * 16 + lr;
    float b2v[E_];
#pragma unroll
    for (int e = 0; e < E_; ++e) b2v[e] = b2[e * DIM_ + d];
#pragma unroll
    for (int rt = 0; rt < 4; ++rt) {
#pragma unroll
      for (int i = 0; i < 8; ++i) {
        int row = rt * 16 + lh * 8 + i;
        float v = acc[rt][ctl][i];
#pragma unroll
        for (int e = 0; e < E_; ++e) v += ps[row][e] * b2v[e];
        long tok = n0 + row;
        long bimg = tok >> 10, pix = tok & 1023;
        out[(bimg * DIM_ + d) * PIX_ + pix] = v;
      }
    }
  }
}

// ---------------- host launcher ----------------
extern "C" void kernel_launch(void* const* d_in, const int* in_sizes, int n_in,
                              void* d_out, int out_size, void* d_ws, size_t ws_size,
                              hipStream_t stream) {
  const float* x    = (const float*)d_in[0];
  const float* w    = (const float*)d_in[1];
  const float* fmu  = (const float*)d_in[2];
  const float* frho = (const float*)d_in[3];
  const float* tmu  = (const float*)d_in[4];
  const float* trho = (const float*)d_in[5];
  const float* cmu  = (const float*)d_in[6];
  const float* crho = (const float*)d_in[7];
  const float* temp = (const float*)d_in[8];
  const float* ef   = (const float*)d_in[9];
  const float* et   = (const float*)d_in[10];
  const float* ec   = (const float*)d_in[11];
  const float* W1   = (const float*)d_in[12];
  const float* b1   = (const float*)d_in[13];
  const float* W2   = (const float*)d_in[14];
  const float* b2   = (const float*)d_in[15];

  float* out   = (float*)d_out;          // [16][256][1024]
  float* kl    = out + 4194304;          // scalar
  float* probs = out + 4194305;          // [N][E]

  char* ws = (char*)d_ws;                // ~17.3 MB total
  _Float16* xh    = (_Float16*)(ws);                 // 8,388,608 B : [N][256] f16
  _Float16* W1t   = (_Float16*)(ws + 8388608);       // 4,194,304 B : [E][HID][DIM] f16
  _Float16* W2t   = (_Float16*)(ws + 12582912);      // 4,194,304 B : [E][DIM][HID] f16
  float*    fw    = (float*)(ws + 16777216);         //   131,072 B
  float*    tw    = (float*)(ws + 16908288);         //   393,216 B
  float*    cw    = (float*)(ws + 17301504);         //     8,192 B
  float*    tcomb = (float*)(ws + 17309696);         //     8,192 B

  sample_weights<<<520, 256, 0, stream>>>(fmu, frho, ef, tmu, trho, et, cmu, crho, ec, fw, tw, cw);

  dim3 tb(32, 8);
  // x: per image [256][1024] -> xh [1024][256]
  transpose_f32_f16<<<dim3(1024 / 32, 256 / 32, 16), tb, 0, stream>>>(x, xh, 256, 1024);
  // W1: per expert [256][1024] -> [1024][256]
  transpose_f32_f16<<<dim3(1024 / 32, 256 / 32, 8), tb, 0, stream>>>(W1, W1t, 256, 1024);
  // W2: per expert [1024][256] -> [256][1024]
  transpose_f32_f16<<<dim3(256 / 32, 1024 / 32, 8), tb, 0, stream>>>(W2, W2t, 1024, 256);

  tcomb_kernel<<<16, 128, 0, stream>>>(w, tw, tcomb);
  router_kernel<<<NTOK_ / 16, 128, 0, stream>>>(xh, fw, cw, tcomb, temp, probs);
  kl_kernel<<<1, 256, 0, stream>>>(fmu, frho, tmu, trho, cmu, crho, kl);

  moe_expert_kernel<<<NTOK_ / MT, 256, 0, stream>>>(xh, W1t, W2t, b1, b2, probs, out);
}
